// MultiHead_39195871543512
// MI455X (gfx1250) — compile-verified, matrix operand order checked by reading, not run on hardware
//
#include <hip/hip_runtime.h>

typedef __bf16 bf16;
typedef bf16  v16bf __attribute__((ext_vector_type(16)));
typedef bf16  v8bf  __attribute__((ext_vector_type(8)));
typedef float v8f   __attribute__((ext_vector_type(8)));
typedef float v4f   __attribute__((ext_vector_type(4)));

#define DEV static __device__ __forceinline__

constexpr int B = 4, T = 2048, F = 1024, H = 16, HD = 64;
constexpr int BH = B * H;

DEV v8f wmma_bf16(v16bf a, v16bf b, v8f c) {
  return __builtin_amdgcn_wmma_f32_16x16x32_bf16(false, a, false, b, (short)0, c,
                                                 false, false);
}

// Build a 16-bit WMMA operand (A or B) for one lane from a K-innermost row.
// Lane-half ha owns K = {8ha..8ha+7} and {16+8ha..23+8ha} of the 32-wide K
// window, packed in order -> two contiguous 8xbf16 runs (2x global_load_b128).
DEV v16bf ldop_bf16(const bf16* base, int ha) {
  v8bf lo = *(const v8bf*)(base + 8 * ha);
  v8bf hi = *(const v8bf*)(base + 16 + 8 * ha);
  v16bf r;
#pragma unroll
  for (int i = 0; i < 8; ++i) { r[i] = lo[i]; r[8 + i] = hi[i]; }
  return r;
}

DEV v16bf ldop_f32(const float* base, int ha) {
  v4f a0 = *(const v4f*)(base + 8 * ha);
  v4f a1 = *(const v4f*)(base + 8 * ha + 4);
  v4f b0 = *(const v4f*)(base + 16 + 8 * ha);
  v4f b1 = *(const v4f*)(base + 20 + 8 * ha);
  v16bf r;
#pragma unroll
  for (int i = 0; i < 4; ++i) {
    r[i]      = (bf16)a0[i];
    r[4 + i]  = (bf16)a1[i];
    r[8 + i]  = (bf16)b0[i];
    r[12 + i] = (bf16)b1[i];
  }
  return r;
}

// ---------------------------------------------------------------------------
// Kernel 0: one-time weight transposes to bf16, K-innermost.
//   wqT/wkT/wvT[h*HD+d][f] = w[h][f][d]   woT[n][k] = wo[k][n]
// ---------------------------------------------------------------------------
__global__ __launch_bounds__(256) void prep_kernel(
    const float* __restrict__ wq, const float* __restrict__ wk,
    const float* __restrict__ wv, const float* __restrict__ wo,
    bf16* __restrict__ wqT, bf16* __restrict__ wkT, bf16* __restrict__ wvT,
    bf16* __restrict__ woT) {
  const int z = blockIdx.z;
  const size_t stride = (size_t)gridDim.x * blockDim.x;
  size_t i0 = (size_t)blockIdx.x * blockDim.x + threadIdx.x;
  if (z < 3) {
    const float* w = (z == 0) ? wq : (z == 1) ? wk : wv;
    bf16* wT       = (z == 0) ? wqT : (z == 1) ? wkT : wvT;
    for (size_t i = i0; i < (size_t)H * HD * F; i += stride) {
      size_t f = i % F, hd = i / F, d = hd % HD, h = hd / HD;
      wT[i] = (bf16)w[(h * F + f) * HD + d];
    }
  } else {
    for (size_t i = i0; i < (size_t)F * F; i += stride) {
      size_t k = i % F, n = i / F;
      woT[i] = (bf16)wo[k * F + n];
    }
  }
}

// ---------------------------------------------------------------------------
// Kernel 1: projections as one GEMM [B*T,F] x [F,F]; 64-wide N slab == head.
// One wave -> 32x64 tile (2 M-subtiles reuse each B operand -> 8 wmma/K-step).
//   q,k heads stored row-major [bh][t][d]; v stored transposed [bh][d][t].
// ---------------------------------------------------------------------------
__global__ __launch_bounds__(128) void proj_kernel(
    const float* __restrict__ q, const float* __restrict__ k,
    const float* __restrict__ v, const bf16* __restrict__ wqT,
    const bf16* __restrict__ wkT, const bf16* __restrict__ wvT,
    bf16* __restrict__ qh, bf16* __restrict__ kh, bf16* __restrict__ vT) {
  const int tid = threadIdx.x;
  const int wave = tid >> 5, lane = tid & 31;
  const int half = lane >> 4, r = lane & 15;

  const int m0 = (blockIdx.x * 4 + wave) * 32;  // 32-row slab (within one b)
  const int h = blockIdx.y;                      // head == 64-wide N slab
  const int which = blockIdx.z;

  const float* src = (which == 0) ? q : (which == 1) ? k : v;
  const bf16* wT   = (which == 0) ? wqT : (which == 1) ? wkT : wvT;

  const float* aRow0 = src + (size_t)(m0 + r) * F;
  const float* aRow1 = src + (size_t)(m0 + 16 + r) * F;
  const bf16* bBase = wT + (size_t)h * HD * F;

  v8f acc0[4] = {}, acc1[4] = {};
  for (int k0 = 0; k0 < F; k0 += 32) {
    if (k0 + 32 < F) {
      __builtin_prefetch(aRow0 + k0 + 32, 0, 1);
      __builtin_prefetch(aRow1 + k0 + 32, 0, 1);
    }
    v16bf a0 = ldop_f32(aRow0 + k0, half);
    v16bf a1 = ldop_f32(aRow1 + k0, half);
#pragma unroll
    for (int dt = 0; dt < 4; ++dt) {
      v16bf bm = ldop_bf16(bBase + (size_t)(dt * 16 + r) * F + k0, half);
      acc0[dt] = wmma_bf16(a0, bm, acc0[dt]);
      acc1[dt] = wmma_bf16(a1, bm, acc1[dt]);
    }
  }

  const int b = m0 / T, t0 = m0 % T;
  const int bh = b * H + h;
  if (which < 2) {
    bf16* out = ((which == 0) ? qh : kh) + ((size_t)bh * T + t0) * HD;
#pragma unroll
    for (int dt = 0; dt < 4; ++dt)
#pragma unroll
      for (int j = 0; j < 8; ++j) {
        out[(size_t)(j + 8 * half) * HD + dt * 16 + r]      = (bf16)acc0[dt][j];
        out[(size_t)(16 + j + 8 * half) * HD + dt * 16 + r] = (bf16)acc1[dt][j];
      }
  } else {
    bf16* out = vT + (size_t)bh * HD * T + t0;  // [d][t]
#pragma unroll
    for (int dt = 0; dt < 4; ++dt)
#pragma unroll
      for (int j = 0; j < 8; ++j) {
        out[(size_t)(dt * 16 + r) * T + j + 8 * half]      = (bf16)acc0[dt][j];
        out[(size_t)(dt * 16 + r) * T + 16 + j + 8 * half] = (bf16)acc1[dt][j];
      }
  }
}

// ---------------------------------------------------------------------------
// Kernel 2: flash attention; one wave per 16-row query tile, 64-key chunks.
// ---------------------------------------------------------------------------
__global__ __launch_bounds__(128) void attn_kernel(
    const bf16* __restrict__ qh, const bf16* __restrict__ kh,
    const bf16* __restrict__ vT, bf16* __restrict__ bt) {
  // P scratch per wave in exact A-operand order: [lane][32] bf16 (2 windows).
  __shared__ alignas(64) bf16 lP[4][32 * 32];

  const int tid = threadIdx.x;
  const int wave = tid >> 5, lane = tid & 31;
  const int half = lane >> 4, r = lane & 15;

  const int qt = blockIdx.x * 4 + wave;
  const int bh = blockIdx.y;
  const int b = bh / H, h = bh % H;

  const bf16* Q = qh + (size_t)bh * T * HD;
  const bf16* K = kh + (size_t)bh * T * HD;
  const bf16* Vt = vT + (size_t)bh * HD * T;  // [d][t]
  const int q0 = qt * 16;

  const bf16* qp = Q + (size_t)(q0 + r) * HD;
  v16bf aq0 = ldop_bf16(qp, half);       // d = 0..31
  v16bf aq1 = ldop_bf16(qp + 32, half);  // d = 32..63

  v8f o0 = {}, o1 = {}, o2 = {}, o3 = {};
  float m[8], l[8];
#pragma unroll
  for (int j = 0; j < 8; ++j) { m[j] = -3.0e38f; l[j] = 0.f; }

  const float scale = 0.03125f;  // 1/(sqrt(1024)+1e-8)
  bf16* myP = &lP[wave][0];
  const int hr = r >> 3;  // lane-half owning key column r in A-layout

  for (int kc = 0; kc < T / 64; ++kc) {
    const int kt0 = kc * 64;
    const bf16* kp = K + (size_t)(kt0 + r) * HD;
    if (kt0 + 64 < T) __builtin_prefetch(kp + (size_t)64 * HD, 0, 1);

    v8f s0 = {}, s1 = {}, s2 = {}, s3 = {};
    s0 = wmma_bf16(aq0, ldop_bf16(kp, half), s0);
    s0 = wmma_bf16(aq1, ldop_bf16(kp + 32, half), s0);
    kp += (size_t)16 * HD;
    s1 = wmma_bf16(aq0, ldop_bf16(kp, half), s1);
    s1 = wmma_bf16(aq1, ldop_bf16(kp + 32, half), s1);
    kp += (size_t)16 * HD;
    s2 = wmma_bf16(aq0, ldop_bf16(kp, half), s2);
    s2 = wmma_bf16(aq1, ldop_bf16(kp + 32, half), s2);
    kp += (size_t)16 * HD;
    s3 = wmma_bf16(aq0, ldop_bf16(kp, half), s3);
    s3 = wmma_bf16(aq1, ldop_bf16(kp + 32, half), s3);

    // Online softmax over 64 keys. C/D layout: row = j+8*half, col = lane&15.
#pragma unroll
    for (int j = 0; j < 8; ++j) {
      float x0 = s0[j] * scale, x1 = s1[j] * scale;
      float x2 = s2[j] * scale, x3 = s3[j] * scale;
      float mv = fmaxf(fmaxf(x0, x1), fmaxf(x2, x3));
      mv = fmaxf(mv, __shfl_xor(mv, 1));
      mv = fmaxf(mv, __shfl_xor(mv, 2));
      mv = fmaxf(mv, __shfl_xor(mv, 4));
      mv = fmaxf(mv, __shfl_xor(mv, 8));
      float mn = fmaxf(m[j], mv);
      float alpha = __expf(m[j] - mn);
      float p0 = __expf(x0 - mn), p1 = __expf(x1 - mn);
      float p2 = __expf(x2 - mn), p3 = __expf(x3 - mn);
      float rs = (p0 + p1) + (p2 + p3);
      rs += __shfl_xor(rs, 1);
      rs += __shfl_xor(rs, 2);
      rs += __shfl_xor(rs, 4);
      rs += __shfl_xor(rs, 8);
      l[j] = l[j] * alpha + rs;
      m[j] = mn;
      o0[j] *= alpha; o1[j] *= alpha; o2[j] *= alpha; o3[j] *= alpha;
      // scatter P into A-operand order: window0 = keys 0..31, window1 = 32..63
      bf16* rowp = myP + ((j + 8 * half) + 16 * hr) * 32;
      rowp[(r & 7)]      = (bf16)p0;  // window0, key r
      rowp[8 + (r & 7)]  = (bf16)p1;  // window0, key 16+r
      rowp[16 + (r & 7)] = (bf16)p2;  // window1, key r
      rowp[24 + (r & 7)] = (bf16)p3;  // window1, key 16+r
    }

    // Same-wave LDS ops are in-order: read back as contiguous vectors.
    v16bf ap0 = *(const v16bf*)(myP + lane * 32);
    v16bf ap1 = *(const v16bf*)(myP + lane * 32 + 16);

#pragma unroll
    for (int dt = 0; dt < 4; ++dt) {
      const bf16* vp = Vt + (size_t)(dt * 16 + r) * T + kt0;
      v16bf bv0 = ldop_bf16(vp, half);
      v16bf bv1 = ldop_bf16(vp + 32, half);
      if (dt == 0)      { o0 = wmma_bf16(ap0, bv0, o0); o0 = wmma_bf16(ap1, bv1, o0); }
      else if (dt == 1) { o1 = wmma_bf16(ap0, bv0, o1); o1 = wmma_bf16(ap1, bv1, o1); }
      else if (dt == 2) { o2 = wmma_bf16(ap0, bv0, o2); o2 = wmma_bf16(ap1, bv1, o2); }
      else              { o3 = wmma_bf16(ap0, bv0, o3); o3 = wmma_bf16(ap1, bv1, o3); }
    }
  }

  bf16* out = bt + ((size_t)b * T + q0) * F + (size_t)h * HD;
#pragma unroll
  for (int j = 0; j < 8; ++j) {
    float inv = 1.f / l[j];
    int row = j + 8 * half;
    out[(size_t)row * F + 0 + r]  = (bf16)(o0[j] * inv);
    out[(size_t)row * F + 16 + r] = (bf16)(o1[j] * inv);
    out[(size_t)row * F + 32 + r] = (bf16)(o2[j] * inv);
    out[(size_t)row * F + 48 + r] = (bf16)(o3[j] * inv);
  }
}

// ---------------------------------------------------------------------------
// Kernel 3: output projection. One wave -> 32x64 tile of out = bt @ woT^T.
// ---------------------------------------------------------------------------
__global__ __launch_bounds__(128) void oproj_kernel(
    const bf16* __restrict__ bt, const bf16* __restrict__ woT,
    float* __restrict__ out) {
  const int tid = threadIdx.x;
  const int wave = tid >> 5, lane = tid & 31;
  const int half = lane >> 4, r = lane & 15;

  const int m0 = (blockIdx.x * 4 + wave) * 32;  // 32-row slab
  const int n0 = blockIdx.y * 64;               // 64-wide N slab

  const bf16* aRow0 = bt + (size_t)(m0 + r) * F;
  const bf16* aRow1 = bt + (size_t)(m0 + 16 + r) * F;

  v8f acc0[4] = {}, acc1[4] = {};
  for (int k0 = 0; k0 < F; k0 += 32) {
    if (k0 + 32 < F) {
      __builtin_prefetch(aRow0 + k0 + 32, 0, 1);
      __builtin_prefetch(aRow1 + k0 + 32, 0, 1);
    }
    v16bf a0 = ldop_bf16(aRow0 + k0, half);
    v16bf a1 = ldop_bf16(aRow1 + k0, half);
#pragma unroll
    for (int dt = 0; dt < 4; ++dt) {
      v16bf bm = ldop_bf16(woT + (size_t)(n0 + dt * 16 + r) * F + k0, half);
      acc0[dt] = wmma_bf16(a0, bm, acc0[dt]);
      acc1[dt] = wmma_bf16(a1, bm, acc1[dt]);
    }
  }

  float* o = out + (size_t)m0 * F + n0;
#pragma unroll
  for (int dt = 0; dt < 4; ++dt)
#pragma unroll
    for (int j = 0; j < 8; ++j) {
      o[(size_t)(j + 8 * half) * F + dt * 16 + r]      = acc0[dt][j];
      o[(size_t)(16 + j + 8 * half) * F + dt * 16 + r] = acc1[dt][j];
    }
}

// ---------------------------------------------------------------------------
extern "C" void kernel_launch(void* const* d_in, const int* in_sizes, int n_in,
                              void* d_out, int out_size, void* d_ws,
                              size_t ws_size, hipStream_t stream) {
  const float* q  = (const float*)d_in[0];
  const float* k  = (const float*)d_in[1];
  const float* v  = (const float*)d_in[2];
  const float* wq = (const float*)d_in[3];
  const float* wk = (const float*)d_in[4];
  const float* wv = (const float*)d_in[5];
  const float* wo = (const float*)d_in[6];

  const size_t perHead = (size_t)BH * T * HD;  // 8.4M elements
  const size_t wSz = (size_t)H * HD * F;       // 1.05M elements
  bf16* qh  = (bf16*)d_ws;
  bf16* kh  = qh + perHead;
  bf16* vT  = kh + perHead;            // [bh][d][t]
  bf16* bt  = vT + perHead;            // [B*T][F]
  bf16* wqT = bt + (size_t)B * T * F;
  bf16* wkT = wqT + wSz;
  bf16* wvT = wkT + wSz;
  bf16* woT = wvT + wSz;               // [F][F]

  prep_kernel<<<dim3(512, 1, 4), 256, 0, stream>>>(wq, wk, wv, wo, wqT, wkT,
                                                   wvT, woT);
  proj_kernel<<<dim3(B * T / 128, H, 3), 128, 0, stream>>>(q, k, v, wqT, wkT,
                                                           wvT, qh, kh, vT);
  attn_kernel<<<dim3(T / 64, BH), 128, 0, stream>>>(qh, kh, vT, bt);
  oproj_kernel<<<dim3(B * T / 128, F / 64), 128, 0, stream>>>(bt, woT,
                                                              (float*)d_out);
}
